// BidirectionalCfC_11922829214115
// MI455X (gfx1250) — compile-verified, matrix operand order checked by reading, not run on hardware
//
#include <hip/hip_runtime.h>
#include <math.h>

// ---------------- types ----------------
typedef __attribute__((ext_vector_type(16))) __bf16 bfvec16;
typedef __attribute__((ext_vector_type(8)))  float  v8f;

#define NB   32        // batch
#define NT   2048      // time
#define TOK  65536     // NB*NT
#define NH   256       // hidden
#define L0   20        // INTER
#define L1   12        // CMD
#define L2   32        // OUT

// scan parameter pack offsets (floats), per direction; 3 gates (ta+tb folded)
#define P_W0H 0                      // 3*20*20 = 1200
#define P_W1X 1200                   // 3*20*12 = 720
#define P_W1H 1920                   // 3*12*12 = 432
#define P_B1  2352                   // 3*12    = 36
#define P_W2X 2388                   // 3*12*32 = 1152
#define P_W2H 3540                   // 3*32*32 = 3072
#define P_B2  6612                   // 3*32    = 96
#define P_SZ  6708

// ---------------- fast activations (v_exp_f32 + v_rcp_f32) ----------------
__device__ __forceinline__ float fsigmoid(float x) {
  return __builtin_amdgcn_rcpf(1.f + __expf(-x));
}
__device__ __forceinline__ float ftanh(float x) {
  return 2.f * __builtin_amdgcn_rcpf(1.f + __expf(-2.f * x)) - 1.f;
}
__device__ __forceinline__ float fsilu(float x) { return x * fsigmoid(x); }

// ---------------- WMMA fragment packing ----------------
// B-matrix fragment layout (wave32, 16-bit, 32x16 tile): lane=(khalf<<4)|n15,
// element i holds K = ((j>>2)<<4)+((j&3)<<1)+(i&1)+(khalf<<3), j=i>>1.
// Packed order: [tileN][kBlock][lane][16] -> two coalesced b128 loads per lane.
__device__ __forceinline__ size_t frag_off(int k, int n, int K) {
  const int k0   = k >> 5;
  const int kr   = k & 31;
  const int half = (kr >> 3) & 1;
  const int hi   = kr >> 4;
  const int low3 = kr & 7;
  const int j    = (hi << 2) | (low3 >> 1);
  const int i    = (j << 1) | (low3 & 1);
  const int lane = (half << 4) | (n & 15);
  return (size_t)(((n >> 4) * (K >> 5) + k0) * 512 + lane * 16 + i);
}

__global__ void pack_w_bf16(const float* __restrict__ W, __bf16* __restrict__ P,
                            int K, int N) {
  int idx = blockIdx.x * blockDim.x + threadIdx.x;
  if (idx >= K * N) return;
  int k = idx / N, n = idx % N;
  P[frag_off(k, n, K)] = (__bf16)W[idx];
}

// ---------------- WMMA 16x16 tile, 2 accumulator chains ----------------
__device__ __forceinline__ v8f wmma_tile_bf16(const __bf16* __restrict__ A,
                                              const __bf16* __restrict__ Wp,
                                              int K, int tileM, int lane) {
  const int half = lane >> 4;
  const int lm   = lane & 15;
  v8f c0 = {0.f,0.f,0.f,0.f,0.f,0.f,0.f,0.f};
  v8f c1 = {0.f,0.f,0.f,0.f,0.f,0.f,0.f,0.f};
  const __bf16* Ap = A + (size_t)(tileM + lm) * K;
  const __bf16* Bq = Wp + lane * 16;
  for (int k0 = 0; k0 < K; k0 += 64) {
    bfvec16 a0, a1;
#pragma unroll
    for (int i = 0; i < 16; ++i) {
      const int j = i >> 1;
      const int k = ((j >> 2) << 4) + ((j & 3) << 1) + (i & 1) + (half << 3);
      a0[i] = Ap[k0 + k];
      a1[i] = Ap[k0 + 32 + k];
    }
    const bfvec16 b0 = *(const bfvec16*)(Bq + ((size_t)(k0 >> 5) << 9));
    const bfvec16 b1 = *(const bfvec16*)(Bq + ((size_t)((k0 >> 5) + 1) << 9));
    c0 = __builtin_amdgcn_wmma_f32_16x16x32_bf16(false, a0, false, b0,
                                                 (short)0, c0, false, false);
    c1 = __builtin_amdgcn_wmma_f32_16x16x32_bf16(false, a1, false, b1,
                                                 (short)0, c1, false, false);
  }
  return c0 + c1;
}

// ---------------- generic GEMM: out = act(A[M,K] @ W[K,N] + bias) ----------------
__global__ void __launch_bounds__(128)
gemm_bias_act(const __bf16* __restrict__ A, const __bf16* __restrict__ Wp,
              const float* __restrict__ bias, __bf16* __restrict__ outB,
              int M, int N, int K, int act) {
  const int lane = threadIdx.x & 31;
  const int wave = threadIdx.x >> 5;
  const int tilesN = N >> 4;
  const int tileIdx = blockIdx.x * 4 + wave;
  const int tileM = (tileIdx / tilesN) << 4;
  const int tileN = (tileIdx % tilesN) << 4;
  if (tileM >= M) return;  // wave-uniform
  const __bf16* Wt = Wp + (size_t)(tileN >> 4) * (K >> 5) * 512;
  v8f c = wmma_tile_bf16(A, Wt, K, tileM, lane);
  const int half = lane >> 4, lm = lane & 15;
  const int col = tileN + lm;
  const float bv = bias ? bias[col] : 0.f;
#pragma unroll
  for (int r = 0; r < 8; ++r) {
    float v = c[r] + bv;
    if (act) v = fsilu(v);
    outB[(size_t)(tileM + r + (half << 3)) * N + col] = (__bf16)v;
  }
}

// ---------------- pre-activation projection for the scan ----------------
// A = h_seq [TOK,256], W packed [256,128]: col = dir*64 + gate*20 + j (3 gates,
// ta/tb folded; cols 60..63 per dir unused). Backward direction time-flipped.
__global__ void __launch_bounds__(128)
gemm_pre0(const __bf16* __restrict__ A, const __bf16* __restrict__ Wp,
          const float* __restrict__ bias, float* __restrict__ pre0) {
  const int K = 256;
  const int lane = threadIdx.x & 31;
  const int wave = threadIdx.x >> 5;
  const int tileIdx = blockIdx.x * 4 + wave;
  const int tileM = (tileIdx >> 3) << 4;
  const int tileN = (tileIdx & 7) << 4;
  const __bf16* Wt = Wp + (size_t)(tileN >> 4) * (K >> 5) * 512;
  v8f c = wmma_tile_bf16(A, Wt, K, tileM, lane);
  const int half = lane >> 4, lm = lane & 15;
  const int col = tileN + lm;
  const float bv = bias[col];
  const int d = col >> 6, rem = col & 63;
#pragma unroll
  for (int r = 0; r < 8; ++r) {
    const int token = tileM + r + (half << 3);
    const int b = token >> 11, t = token & (NT - 1);
    const int s = d ? (NT - 1 - t) : t;
    pre0[(((size_t)d * NT + s) * NB + b) * 64 + rem] = c[r] + bv;
  }
}

// ---------------- small prep kernels ----------------
// one (dir,gate) 20-column slice of the fused pre0 weight/bias (fp32 staging)
__global__ void build_pre0_slice(const float* __restrict__ w, const float* __restrict__ mask,
                                 const float* __restrict__ bias, float* __restrict__ pwf,
                                 float* __restrict__ pb, int colOff, int accum) {
  int i = blockIdx.x * blockDim.x + threadIdx.x;   // 256*20
  if (i >= NH * L0) return;
  int k = i / L0, j = i % L0;
  float v = w[k * L0 + j] * mask[k * L0 + j];
  float* dst = &pwf[(size_t)k * 128 + colOff + j];
  if (accum) *dst += v; else *dst = v;
  if (k == 0) {
    if (accum) pb[colOff + j] += bias[j]; else pb[colOff + j] = bias[j];
  }
}

// split masked (in+n,n) weight into x-part / h-part / bias in the scan pack
__global__ void prep_gate(const float* __restrict__ w, const float* __restrict__ mask,
                          const float* __restrict__ bias, float* __restrict__ dstX,
                          float* __restrict__ dstH, float* __restrict__ dstB,
                          int in_dim, int n, int accum) {
  int i = blockIdx.x * blockDim.x + threadIdx.x;
  int total = (in_dim + n) * n;
  if (i < total) {
    int k = i / n, j = i % n;
    float v = w[i] * (mask ? mask[i] : 1.f);
    if (k < in_dim) {
      if (dstX) { if (accum) dstX[k * n + j] += v; else dstX[k * n + j] = v; }
    } else {
      float* d = &dstH[(size_t)(k - in_dim) * n + j];
      if (accum) *d += v; else *d = v;
    }
  } else if (i < total + n && dstB) {
    int j = i - total;
    if (accum) dstB[j] += bias[j]; else dstB[j] = bias[j];
  }
}

// ---------------- input layer (K=9, VALU) ----------------
__global__ void enc_first(const float* __restrict__ x, const float* __restrict__ W0,
                          const float* __restrict__ b0, __bf16* __restrict__ h1) {
  int idx = blockIdx.x * blockDim.x + threadIdx.x;   // TOK*256
  int col = idx & (NH - 1);
  int token = idx >> 8;
  const float* xp = x + token * 9;
  float s = b0[col];
#pragma unroll
  for (int k = 0; k < 9; ++k) s += xp[k] * W0[k * NH + col];
  h1[idx] = (__bf16)fsilu(s);
}

// ---------------- CfC scan: one workgroup per direction, weights+state in LDS ----
__global__ void __launch_bounds__(256)
cfc_scan(const float* __restrict__ pre0, const float* __restrict__ sp,
         __bf16* __restrict__ hcat) {
  const int dir = blockIdx.x;
  const int tid = threadIdx.x;
  __shared__ float W[P_SZ];
  __shared__ float h0[2][NB * L0];
  __shared__ float h1[2][NB * L1];
  __shared__ float h2[2][NB * L2];
  for (int i = tid; i < P_SZ; i += 256) W[i] = sp[dir * P_SZ + i];
  for (int i = tid; i < NB * L0; i += 256) h0[0][i] = 0.f;
  for (int i = tid; i < NB * L1; i += 256) h1[0][i] = 0.f;
  for (int i = tid; i < NB * L2; i += 256) h2[0][i] = 0.f;
  __syncthreads();
  int cur = 0;
  for (int t = 0; t < NT; ++t) {
    const int nxt = cur ^ 1;
    const float* pre = pre0 + ((size_t)dir * NT + t) * (NB * 64);
    // prefetch next step's pre-activation block (NB*64*4 = 8 KB)
    if (t + 1 < NT && tid < 32)
      __builtin_prefetch((const void*)(pre + NB * 64 + tid * 64), 0, 0);
    // ---- layer 0 (x-part + bias already in pre0; gates: ff1, ff2, ta+tb) ----
    for (int i = tid; i < NB * L0; i += 256) {
      const int b = i / L0, j = i % L0;
      const float* p = pre + b * 64;
      float s1 = p[j], s2 = p[20 + j], sg = p[40 + j];
      const float* h = &h0[cur][b * L0];
      for (int k = 0; k < L0; ++k) {
        const float hv = h[k];
        s1 += hv * W[P_W0H + k * L0 + j];
        s2 += hv * W[P_W0H + 400 + k * L0 + j];
        sg += hv * W[P_W0H + 800 + k * L0 + j];
      }
      const float g = fsigmoid(sg);
      h0[nxt][i] = ftanh(s1) * (1.f - g) + g * ftanh(s2);
    }
    __syncthreads();
    // ---- layer 1 ----
    for (int i = tid; i < NB * L1; i += 256) {
      const int b = i / L1, j = i % L1;
      const float* xin = &h0[nxt][b * L0];
      const float* h = &h1[cur][b * L1];
      float s1 = W[P_B1 + j], s2 = W[P_B1 + 12 + j], sg = W[P_B1 + 24 + j];
      for (int k = 0; k < L0; ++k) {
        const float xv = xin[k];
        s1 += xv * W[P_W1X + k * L1 + j];
        s2 += xv * W[P_W1X + 240 + k * L1 + j];
        sg += xv * W[P_W1X + 480 + k * L1 + j];
      }
      for (int k = 0; k < L1; ++k) {
        const float hv = h[k];
        s1 += hv * W[P_W1H + k * L1 + j];
        s2 += hv * W[P_W1H + 144 + k * L1 + j];
        sg += hv * W[P_W1H + 288 + k * L1 + j];
      }
      const float g = fsigmoid(sg);
      h1[nxt][i] = ftanh(s1) * (1.f - g) + g * ftanh(s2);
    }
    __syncthreads();
    // ---- layer 2 (+ emit output) ----
    for (int i = tid; i < NB * L2; i += 256) {
      const int b = i >> 5, j = i & 31;
      const float* xin = &h1[nxt][b * L1];
      const float* h = &h2[cur][b * L2];
      float s1 = W[P_B2 + j], s2 = W[P_B2 + 32 + j], sg = W[P_B2 + 64 + j];
      for (int k = 0; k < L1; ++k) {
        const float xv = xin[k];
        s1 += xv * W[P_W2X + k * L2 + j];
        s2 += xv * W[P_W2X + 384 + k * L2 + j];
        sg += xv * W[P_W2X + 768 + k * L2 + j];
      }
      for (int k = 0; k < L2; ++k) {
        const float hv = h[k];
        s1 += hv * W[P_W2H + k * L2 + j];
        s2 += hv * W[P_W2H + 1024 + k * L2 + j];
        sg += hv * W[P_W2H + 2048 + k * L2 + j];
      }
      const float g = fsigmoid(sg);
      const float out = ftanh(s1) * (1.f - g) + g * ftanh(s2);
      h2[nxt][i] = out;
      const int tt = dir ? (NT - 1 - t) : t;
      hcat[((size_t)b * NT + tt) * 64 + dir * 32 + j] = (__bf16)out;
    }
    __syncthreads();
    cur = nxt;
  }
}

// ---------------- final decoder layer (N=2, VALU) ----------------
__global__ void dec_final(const __bf16* __restrict__ h, const float* __restrict__ W,
                          const float* __restrict__ bias, float* __restrict__ out) {
  int token = blockIdx.x * blockDim.x + threadIdx.x;
  if (token >= TOK) return;
  const __bf16* hp = h + (size_t)token * 64;
  float s0 = bias[0], s1 = bias[1];
#pragma unroll 8
  for (int k = 0; k < 64; ++k) {
    const float hv = (float)hp[k];
    s0 += hv * W[k * 2 + 0];
    s1 += hv * W[k * 2 + 1];
  }
  out[token] = s0;
  out[TOK + token] = s1;
}

// ================= host side =================
extern "C" void kernel_launch(void* const* d_in, const int* in_sizes, int n_in,
                              void* d_out, int out_size, void* d_ws, size_t ws_size,
                              hipStream_t stream) {
  (void)in_sizes; (void)n_in; (void)out_size; (void)ws_size;
  auto F = [&](int i) { return (const float*)d_in[i]; };
  // top-level dict order; nested param dicts flatten by sorted key:
  // ff1_b, ff1_w, ff2_b, ff2_w, ta_b, ta_w, tb_b, tb_w
  const float* x     = F(0);
  const float* encW0 = F(1);  const float* encB0 = F(2);
  const float* encW1 = F(3);  const float* encB1 = F(4);
  const float* encW2 = F(5);  const float* encB2 = F(6);
  const int cfcBase[2]  = {7, 31};
  const int maskBase[2] = {65, 68};
  const float* combW0 = F(55); const float* combB0 = F(56);
  const float* combW1 = F(57); const float* combB1 = F(58);
  const float* decW0  = F(59); const float* decB0  = F(60);
  const float* decW1  = F(61); const float* decB1  = F(62);
  const float* decW2  = F(63); const float* decB2  = F(64);
  float* out = (float*)d_out;

  // workspace carve-up
  char* ws = (char*)d_ws;
  size_t off = 0;
  auto carve = [&](size_t bytes) {
    void* p = ws + off;
    off = (off + bytes + 255) & ~(size_t)255;
    return p;
  };
  __bf16* X1     = (__bf16*)carve((size_t)TOK * NH * 2);
  __bf16* X2     = (__bf16*)carve((size_t)TOK * NH * 2);
  float*  pre0   = (float*) carve((size_t)2 * NT * NB * 64 * 4);
  __bf16* hcat   = (__bf16*)carve((size_t)TOK * 64 * 2);
  __bf16* wEnc1  = (__bf16*)carve((size_t)NH * NH * 2);
  __bf16* wEnc2  = (__bf16*)carve((size_t)NH * NH * 2);
  __bf16* wComb0 = (__bf16*)carve((size_t)64 * NH * 2);
  __bf16* wComb1 = (__bf16*)carve((size_t)NH * NH * 2);
  __bf16* wDec0  = (__bf16*)carve((size_t)NH * 128 * 2);
  __bf16* wDec1  = (__bf16*)carve((size_t)128 * 64 * 2);
  float*  pre0Wf = (float*) carve((size_t)NH * 128 * 4);   // fp32 staging
  __bf16* pre0Wp = (__bf16*)carve((size_t)NH * 128 * 2);   // packed bf16
  float*  pre0B  = (float*) carve(128 * 4);
  float*  sp     = (float*) carve((size_t)2 * P_SZ * 4);

  // --- weight conversion + fragment packing ---
  auto pack = [&](const float* s, __bf16* d, int K, int N) {
    pack_w_bf16<<<(K * N + 255) / 256, 256, 0, stream>>>(s, d, K, N);
  };
  pack(encW1,  wEnc1,  NH, NH);
  pack(encW2,  wEnc2,  NH, NH);
  pack(combW0, wComb0, 64, NH);
  pack(combW1, wComb1, NH, NH);
  pack(decW0,  wDec0,  NH, 128);
  pack(decW1,  wDec1,  128, 64);

  // --- fused pre0 projection (ta+tb folded) + scan parameter packs ---
  for (int d = 0; d < 2; ++d) {
    const int cb = cfcBase[d];
    float* spd = sp + d * P_SZ;
    for (int g = 0; g < 4; ++g) {
      const int slot = (g < 3) ? g : 2;      // fold tb into ta slot
      const int acc = (g == 3) ? 1 : 0;
      build_pre0_slice<<<(NH * L0 + 255) / 256, 256, 0, stream>>>(
          F(cb + 2 * g + 1), F(maskBase[d]), F(cb + 2 * g),
          pre0Wf, pre0B, d * 64 + slot * 20, acc);
      prep_gate<<<((276 * L0 + L0) + 255) / 256, 256, 0, stream>>>(
          F(cb + 2 * g + 1), F(maskBase[d]), nullptr,
          nullptr, spd + P_W0H + slot * 400, nullptr, 256, L0, acc);
      prep_gate<<<((32 * L1 + L1) + 255) / 256, 256, 0, stream>>>(
          F(cb + 8 + 2 * g + 1), F(maskBase[d] + 1), F(cb + 8 + 2 * g),
          spd + P_W1X + slot * 240, spd + P_W1H + slot * 144,
          spd + P_B1 + slot * 12, L0, L1, acc);
      prep_gate<<<((44 * L2 + L2) + 255) / 256, 256, 0, stream>>>(
          F(cb + 16 + 2 * g + 1), F(maskBase[d] + 2), F(cb + 16 + 2 * g),
          spd + P_W2X + slot * 384, spd + P_W2H + slot * 1024,
          spd + P_B2 + slot * 32, L1, L2, acc);
    }
  }
  pack(pre0Wf, pre0Wp, NH, 128);

  // --- encoder ---
  enc_first<<<(TOK * NH) / 256, 256, 0, stream>>>(x, encW0, encB0, X1);
  gemm_bias_act<<<(TOK / 16) * (NH / 16) / 4, 128, 0, stream>>>(
      X1, wEnc1, encB1, X2, TOK, NH, NH, 1);
  gemm_bias_act<<<(TOK / 16) * (NH / 16) / 4, 128, 0, stream>>>(
      X2, wEnc2, encB2, X1, TOK, NH, NH, 0);

  // --- pre-activations for both scan directions (3 fused gates each) ---
  gemm_pre0<<<(TOK / 16) * 8 / 4, 128, 0, stream>>>(X1, pre0Wp, pre0B, pre0);

  // --- bidirectional CfC scan ---
  cfc_scan<<<2, 256, 0, stream>>>(pre0, sp, hcat);

  // --- combiner ---
  gemm_bias_act<<<(TOK / 16) * (NH / 16) / 4, 128, 0, stream>>>(
      hcat, wComb0, combB0, X2, TOK, NH, 64, 1);
  gemm_bias_act<<<(TOK / 16) * (NH / 16) / 4, 128, 0, stream>>>(
      X2, wComb1, combB1, X1, TOK, NH, NH, 0);

  // --- decoder ---
  gemm_bias_act<<<(TOK / 16) * (128 / 16) / 4, 128, 0, stream>>>(
      X1, wDec0, decB0, X2, TOK, 128, NH, 1);
  gemm_bias_act<<<(TOK / 16) * (64 / 16) / 4, 128, 0, stream>>>(
      X2, wDec1, decB1, X1, TOK, 64, 128, 1);
  dec_final<<<TOK / 256, 256, 0, stream>>>(X1, decW2, decB2, out);
}
